// TransformerDecoderLayer_2525440770597
// MI455X (gfx1250) — matrix operand order repaired and checked by reading, not compiled, not run on hardware
//
#include <hip/hip_runtime.h>
#include <hip/hip_bf16.h>

// ---------------------------------------------------------------------------
// Types for CDNA5 WMMA (gfx1250, wave32)
// ---------------------------------------------------------------------------
typedef __attribute__((ext_vector_type(16))) __bf16        v16bf;
typedef __attribute__((ext_vector_type(8)))  float         v8f;
typedef __attribute__((ext_vector_type(4)))  unsigned int  uint4v;  // trivial, union-safe
typedef __attribute__((ext_vector_type(2)))  unsigned int  uint2v;

union BF16Frag {
    v16bf          v;
    unsigned short u[16];
    uint4v         q[2];
};
union U8x16 { uint4v q; unsigned short s[8]; };
union U4x16 { uint2v q; unsigned short s[4]; };

__device__ __forceinline__ unsigned short f2bf(float f) {
    unsigned int u = __float_as_uint(f);
    unsigned int r = u + 0x7FFFu + ((u >> 16) & 1u);   // round-to-nearest-even
    return (unsigned short)(r >> 16);
}
__device__ __forceinline__ float bf2f(unsigned short s) {
    return __uint_as_float((unsigned int)s << 16);
}

__device__ __forceinline__ v8f wmma_bf16(const v16bf a, const v16bf b, v8f c) {
    // v_wmma_f32_16x16x32_bf16  D = A(16x32) * B(32x16) + C(16x16 f32)
    return __builtin_amdgcn_wmma_f32_16x16x32_bf16(false, a, false, b, (short)0, c,
                                                   false, false);
}

static constexpr int D_MODEL = 1024;
static constexpr int N_HEADS = 16;
static constexpr int DK      = 64;
static constexpr int BATCH   = 4;
static constexpr int TLEN    = 1024;
static constexpr int SLEN    = 1024;
static constexpr int D_FF    = 4096;

// ---------------------------------------------------------------------------
// One-time fp32 -> bf16 conversion (row-major copy)
// ---------------------------------------------------------------------------
__global__ __launch_bounds__(256)
void cvt_bf16_kernel(const float* __restrict__ in, unsigned short* __restrict__ out, int n)
{
    int i = (blockIdx.x * 256 + threadIdx.x) * 8;
    if (i >= n) return;
    float4 a = *(const float4*)(in + i);
    float4 b = *(const float4*)(in + i + 4);
    U8x16 u;
    u.s[0] = f2bf(a.x); u.s[1] = f2bf(a.y); u.s[2] = f2bf(a.z); u.s[3] = f2bf(a.w);
    u.s[4] = f2bf(b.x); u.s[5] = f2bf(b.y); u.s[6] = f2bf(b.z); u.s[7] = f2bf(b.w);
    *(uint4v*)(out + i) = u.q;
}

// ---------------------------------------------------------------------------
// One-time fused convert + transpose:  W[K][N] f32  ->  WT[N][K] bf16
// 64x64 tiles via padded LDS.  grid = (N/64, K/64)
// ---------------------------------------------------------------------------
#define TPAD 72

__global__ __launch_bounds__(256)
void tcvt_kernel(const float* __restrict__ in, unsigned short* __restrict__ out,
                 int K, int N)
{
    __shared__ alignas(16) unsigned short T[64 * TPAD];
    const int n0 = blockIdx.x * 64;
    const int k0 = blockIdx.y * 64;
    const int tid = threadIdx.x;
    #pragma unroll
    for (int i = 0; i < 4; ++i) {
        int f  = tid + i * 256;              // 1024 float4
        int kr = f >> 4;                     // 16 float4 per row
        int n4 = (f & 15) << 2;
        float4 w = *(const float4*)(in + (size_t)(k0 + kr) * N + n0 + n4);
        T[(n4 + 0) * TPAD + kr] = f2bf(w.x);
        T[(n4 + 1) * TPAD + kr] = f2bf(w.y);
        T[(n4 + 2) * TPAD + kr] = f2bf(w.z);
        T[(n4 + 3) * TPAD + kr] = f2bf(w.w);
    }
    __syncthreads();
    #pragma unroll
    for (int i = 0; i < 2; ++i) {
        int f  = tid + i * 256;              // 512 uint4
        int nr = f >> 3;                     // 8 uint4 per row
        int k8 = (f & 7) << 3;
        *(uint4v*)(out + (size_t)(n0 + nr) * K + k0 + k8) =
            *(const uint4v*)&T[nr * TPAD + k8];
    }
}

// ---------------------------------------------------------------------------
// GEMM:  C[M,N](bf16) = A[M,K](bf16) * WT[N,K](bf16, pre-transposed) + bias(f32)
// 256 threads = 8 waves; block tile 128x128; wave tile 64x32; K-tile 32.
// Staging is pure b128 copies (no conversion); two-stage software pipeline
// with ping-pong LDS (one barrier per K-step); L2 warmed one tile ahead.
// ---------------------------------------------------------------------------
#define LDT 40   // padded LDS row stride in ushorts (80 B: 16B-aligned, 20-bank step)

__global__ __launch_bounds__(256)
void gemm_bias_kernel(const unsigned short* __restrict__ A,
                      const unsigned short* __restrict__ WT,
                      const float* __restrict__ bias,
                      unsigned short* __restrict__ C,
                      int M, int N, int K, int relu)
{
    __shared__ alignas(16) unsigned short As[2][128 * LDT];   // [m][k] bf16
    __shared__ alignas(16) unsigned short BsT[2][128 * LDT];  // [n][k] bf16

    const int tid   = threadIdx.x;
    const int lane  = tid & 31;
    const int wave  = tid >> 5;        // 0..7
    const int waveM = wave >> 2;       // 0..1  -> 64 rows
    const int waveN = wave & 3;        // 0..3  -> 32 cols
    const int hb    = lane >> 4;       // half-wave
    const int l16   = lane & 15;

    const int blockM = blockIdx.y * 128;
    const int blockN = blockIdx.x * 128;

    // cooperative-load coords: tile = 128 rows x 32 bf16 = 512 uint4; 2/thread
    const int ldRow = tid >> 2;                // f>>2
    const int ldQ8  = (tid & 3) << 3;          // (f&3)*8 ushorts

    v8f acc[4][2] = {};
    uint4v aU[2], bU[2];

    auto loadTile = [&](int k0) {
        #pragma unroll
        for (int i = 0; i < 2; ++i) {
            int row = ldRow + i * 64;          // f = tid + i*256
            aU[i] = *(const uint4v*)(A  + (size_t)(blockM + row) * K + k0 + ldQ8);
            bU[i] = *(const uint4v*)(WT + (size_t)(blockN + row) * K + k0 + ldQ8);
        }
    };
    auto storeTile = [&](int buf) {
        #pragma unroll
        for (int i = 0; i < 2; ++i) {
            int row = ldRow + i * 64;
            *(uint4v*)&As [buf][row * LDT + ldQ8] = aU[i];
            *(uint4v*)&BsT[buf][row * LDT + ldQ8] = bU[i];
        }
    };
    auto compute = [&](int buf) {
        BF16Frag afrag[4], bfrag[2];
        #pragma unroll
        for (int mi = 0; mi < 4; ++mi) {
            // A layout: lanes0-15 M=l16 K:0-7|16-23 ; lanes16-31 K:8-15|24-31
            const unsigned short* src =
                &As[buf][(waveM * 64 + mi * 16 + l16) * LDT + (hb ? 8 : 0)];
            afrag[mi].q[0] = *(const uint4v*)(src);
            afrag[mi].q[1] = *(const uint4v*)(src + 16);
        }
        #pragma unroll
        for (int ni = 0; ni < 2; ++ni) {
            // B layout: lane N=l16; lanes0-15 K:0-7|16-23 ; lanes16-31 K:8-15|24-31
            const unsigned short* src =
                &BsT[buf][(waveN * 32 + ni * 16 + l16) * LDT + (hb ? 8 : 0)];
            bfrag[ni].q[0] = *(const uint4v*)(src);
            bfrag[ni].q[1] = *(const uint4v*)(src + 16);
        }
        #pragma unroll
        for (int mi = 0; mi < 4; ++mi)
            #pragma unroll
            for (int ni = 0; ni < 2; ++ni)
                acc[mi][ni] = wmma_bf16(afrag[mi].v, bfrag[ni].v, acc[mi][ni]);
    };

    // ---- pipelined K loop ----
    const int nKt = K >> 5;
    loadTile(0);
    storeTile(0);
    int buf = 0;
    for (int kt = 1; kt < nKt; ++kt) {
        loadTile(kt * 32);                 // next tile's global loads in flight
        if (kt + 1 < nKt) {                // warm L2 one tile further ahead
            int k0n = (kt + 1) * 32;
            __builtin_prefetch((const void*)(A  + (size_t)(blockM + ldRow) * K + k0n + ldQ8), 0, 1);
            __builtin_prefetch((const void*)(WT + (size_t)(blockN + ldRow) * K + k0n + ldQ8), 0, 1);
        }
        __syncthreads();                   // LDS[buf] complete for all waves
        compute(buf);                      // WMMA overlaps outstanding loads
        storeTile(buf ^ 1);                // waits on loadcnt as data is consumed
        buf ^= 1;
    }
    __syncthreads();
    compute(buf);

    // ---- epilogue: D layout lanes0-15 M=v, lanes16-31 M=v+8, N=l16 ----
    #pragma unroll
    for (int mi = 0; mi < 4; ++mi) {
        #pragma unroll
        for (int ni = 0; ni < 2; ++ni) {
            int col = blockN + waveN * 32 + ni * 16 + l16;
            float bv = bias[col];
            int rowb = blockM + waveM * 64 + mi * 16 + hb * 8;
            #pragma unroll
            for (int v = 0; v < 8; ++v) {
                float r = acc[mi][ni][v] + bv;
                if (relu) r = fmaxf(r, 0.0f);
                C[(size_t)(rowb + v) * N + col] = f2bf(r);
            }
        }
    }
}

// ---------------------------------------------------------------------------
// Fused flash attention: grid (T/64, B*H), 128 threads = 4 waves.
// Wave w owns query rows q0+16w .. +15. dk = 64. Key tiles of 64.
// Q/K/V/O layout: [B, seq, H, 64] bf16 (projection output [B*seq, D]).
// ---------------------------------------------------------------------------
#define ATP 72   // padded row stride (144 B: 16B-aligned, 36-bank step)

__global__ __launch_bounds__(128)
void attn_kernel(const unsigned short* __restrict__ Qg,
                 const unsigned short* __restrict__ Kg,
                 const unsigned short* __restrict__ Vg,
                 unsigned short* __restrict__ Og,
                 int Tq, int Sk, int causal)
{
    __shared__ alignas(16) unsigned short Ks[64 * ATP];        // [key][d]
    __shared__ alignas(16) unsigned short Vt[64 * ATP];        // [d][key]
    __shared__ alignas(16) unsigned short Ps[4 * 16 * ATP];    // per-wave [row][key]

    const int tid  = threadIdx.x;
    const int lane = tid & 31;
    const int wave = tid >> 5;     // 0..3
    const int hb   = lane >> 4;
    const int l16  = lane & 15;

    const int q0 = blockIdx.x * 64;
    const int bh = blockIdx.y;
    const int b  = bh / N_HEADS;
    const int h  = bh % N_HEADS;

    // ---- load Q A-fragments once (rows q0+16*wave+l16, d=0..63 -> 2 frags) ----
    const unsigned short* qp =
        Qg + (((size_t)(b * Tq) + (q0 + wave * 16 + l16)) * N_HEADS + h) * DK;
    BF16Frag qfrag[2];
    #pragma unroll
    for (int fgi = 0; fgi < 2; ++fgi) {
        int kb = fgi * 32 + (hb ? 8 : 0);
        qfrag[fgi].q[0] = *(const uint4v*)(qp + kb);
        qfrag[fgi].q[1] = *(const uint4v*)(qp + kb + 16);
    }

    float rm[8], rl[8];
    #pragma unroll
    for (int v = 0; v < 8; ++v) { rm[v] = -1e30f; rl[v] = 0.0f; }
    v8f oacc[4] = {};

    const int nkt = causal ? (q0 / 64 + 1) : (Sk / 64);
    for (int kt = 0; kt < nkt; ++kt) {
        const int kk0 = kt * 64;
        __syncthreads();
        // ---- stage K (row major) and V (transposed) tiles: 64x64 bf16 ----
        #pragma unroll
        for (int i = 0; i < 4; ++i) {
            int f   = tid + i * 128;       // 512 uint4 total
            int key = f >> 3;
            int d8  = (f & 7) << 3;
            size_t base = (((size_t)(b * Sk) + kk0 + key) * N_HEADS + h) * DK + d8;
            *(uint4v*)&Ks[key * ATP + d8] = *(const uint4v*)(Kg + base);
            U8x16 vU; vU.q = *(const uint4v*)(Vg + base);
            #pragma unroll
            for (int j = 0; j < 8; ++j)
                Vt[(d8 + j) * ATP + key] = vU.s[j];
        }
        __syncthreads();

        // ---- scores: S = Q K^T  (4 key sub-tiles x 2 K-dim frags) ----
        v8f sacc[4] = {};
        BF16Frag bfr;
        #pragma unroll
        for (int ns = 0; ns < 4; ++ns) {
            #pragma unroll
            for (int kf = 0; kf < 2; ++kf) {
                const unsigned short* src =
                    &Ks[(ns * 16 + l16) * ATP + kf * 32 + (hb ? 8 : 0)];
                bfr.q[0] = *(const uint4v*)(src);
                bfr.q[1] = *(const uint4v*)(src + 16);
                sacc[ns] = wmma_bf16(qfrag[kf].v, bfr.v, sacc[ns]);
            }
        }
        // scale + causal mask (C layout: N=l16, M=v+8*hb)
        const float scale = 0.125f;    // 1/sqrt(64)
        #pragma unroll
        for (int ns = 0; ns < 4; ++ns) {
            #pragma unroll
            for (int v = 0; v < 8; ++v) {
                float s = sacc[ns][v] * scale;
                if (causal) {
                    int kk = kk0 + ns * 16 + l16;
                    int qq = q0 + wave * 16 + v + hb * 8;
                    if (kk > qq) s = -1e9f;
                }
                sacc[ns][v] = s;
            }
        }
        // ---- online softmax (row reduce over 16 lanes of each half) ----
        float pr[4][8];
        #pragma unroll
        for (int v = 0; v < 8; ++v) {
            float mp = fmaxf(fmaxf(sacc[0][v], sacc[1][v]), fmaxf(sacc[2][v], sacc[3][v]));
            #pragma unroll
            for (int off = 8; off >= 1; off >>= 1)
                mp = fmaxf(mp, __shfl_xor(mp, off, 32));
            float mnew  = fmaxf(rm[v], mp);
            float alpha = __expf(rm[v] - mnew);
            rm[v] = mnew;
            float sp = 0.0f;
            #pragma unroll
            for (int ns = 0; ns < 4; ++ns) {
                float p = __expf(sacc[ns][v] - mnew);
                pr[ns][v] = p;
                sp += p;
            }
            #pragma unroll
            for (int off = 8; off >= 1; off >>= 1)
                sp += __shfl_xor(sp, off, 32);
            rl[v] = rl[v] * alpha + sp;
            #pragma unroll
            for (int ds = 0; ds < 4; ++ds)
                oacc[ds][v] = oacc[ds][v] * alpha;
        }
        // ---- P (C layout) -> per-wave LDS -> A-layout fragments ----
        unsigned short* prow = &Ps[wave * 16 * ATP];
        #pragma unroll
        for (int v = 0; v < 8; ++v) {
            int row = v + hb * 8;
            #pragma unroll
            for (int ns = 0; ns < 4; ++ns)
                prow[row * ATP + ns * 16 + l16] = f2bf(pr[ns][v]);
        }
        // same-wave LDS ops are in-order; compiler inserts s_wait_dscnt
        BF16Frag pfrag[2];
        #pragma unroll
        for (int fgi = 0; fgi < 2; ++fgi) {
            const unsigned short* src = &prow[l16 * ATP + fgi * 32 + (hb ? 8 : 0)];
            pfrag[fgi].q[0] = *(const uint4v*)(src);
            pfrag[fgi].q[1] = *(const uint4v*)(src + 16);
        }
        // ---- ctx += P * V  (4 dk sub-tiles x 2 key-dim frags) ----
        #pragma unroll
        for (int ds = 0; ds < 4; ++ds) {
            #pragma unroll
            for (int kf = 0; kf < 2; ++kf) {
                const unsigned short* src =
                    &Vt[(ds * 16 + l16) * ATP + kf * 32 + (hb ? 8 : 0)];
                bfr.q[0] = *(const uint4v*)(src);
                bfr.q[1] = *(const uint4v*)(src + 16);
                oacc[ds] = wmma_bf16(pfrag[kf].v, bfr.v, oacc[ds]);
            }
        }
    }

    // ---- write O = ctx / rowsum (bf16) ----
    #pragma unroll
    for (int ds = 0; ds < 4; ++ds) {
        #pragma unroll
        for (int v = 0; v < 8; ++v) {
            int row = q0 + wave * 16 + v + hb * 8;
            int d   = ds * 16 + l16;
            Og[(((size_t)(b * Tq) + row) * N_HEADS + h) * DK + d] =
                f2bf(oacc[ds][v] / rl[v]);
        }
    }
}

// ---------------------------------------------------------------------------
// out = LayerNorm(x + a) * g + b   — one workgroup per row of 1024.
// Inputs bf16; fp32 statistics; output bf16 (ws) or fp32 (d_out) via flag.
// ---------------------------------------------------------------------------
__global__ __launch_bounds__(256)
void ln_kernel(const unsigned short* __restrict__ X,
               const unsigned short* __restrict__ Ar,
               const float* __restrict__ g, const float* __restrict__ bta,
               void* __restrict__ out, int out32)
{
    __shared__ float red[16];
    const int tid = threadIdx.x;
    const size_t rb = (size_t)blockIdx.x * D_MODEL;

    U4x16 xu, au;
    xu.q = *(const uint2v*)(X  + rb + tid * 4);
    au.q = *(const uint2v*)(Ar + rb + tid * 4);
    float s0 = bf2f(xu.s[0]) + bf2f(au.s[0]);
    float s1 = bf2f(xu.s[1]) + bf2f(au.s[1]);
    float s2 = bf2f(xu.s[2]) + bf2f(au.s[2]);
    float s3 = bf2f(xu.s[3]) + bf2f(au.s[3]);

    float sum = s0 + s1 + s2 + s3;
    float sq  = s0 * s0 + s1 * s1 + s2 * s2 + s3 * s3;
    #pragma unroll
    for (int off = 16; off >= 1; off >>= 1) {
        sum += __shfl_xor(sum, off, 32);
        sq  += __shfl_xor(sq,  off, 32);
    }
    if ((tid & 31) == 0) { red[tid >> 5] = sum; red[8 + (tid >> 5)] = sq; }
    __syncthreads();
    float tot = 0.0f, totq = 0.0f;
    #pragma unroll
    for (int i = 0; i < 8; ++i) { tot += red[i]; totq += red[8 + i]; }

    const float mu  = tot * (1.0f / D_MODEL);
    const float var = totq * (1.0f / D_MODEL) - mu * mu;
    const float inv = rsqrtf(var + 1e-5f);

    float4 g4 = *(const float4*)(g + tid * 4);
    float4 b4 = *(const float4*)(bta + tid * 4);
    float o0 = (s0 - mu) * inv * g4.x + b4.x;
    float o1 = (s1 - mu) * inv * g4.y + b4.y;
    float o2 = (s2 - mu) * inv * g4.z + b4.z;
    float o3 = (s3 - mu) * inv * g4.w + b4.w;

    if (out32) {
        float4 o; o.x = o0; o.y = o1; o.z = o2; o.w = o3;
        *(float4*)((float*)out + rb + tid * 4) = o;
    } else {
        U4x16 ou;
        ou.s[0] = f2bf(o0); ou.s[1] = f2bf(o1);
        ou.s[2] = f2bf(o2); ou.s[3] = f2bf(o3);
        *(uint2v*)((unsigned short*)out + rb + tid * 4) = ou.q;
    }
}

// ---------------------------------------------------------------------------
// Orchestration
// ---------------------------------------------------------------------------
extern "C" void kernel_launch(void* const* d_in, const int* in_sizes, int n_in,
                              void* d_out, int out_size, void* d_ws, size_t ws_size,
                              hipStream_t stream) {
    (void)in_sizes; (void)n_in; (void)out_size;

    const float* x    = (const float*)d_in[0];
    const float* enc  = (const float*)d_in[1];
    // d_in[2]=src_mask (all ones), d_in[3]=tgt_mask (tril): handled analytically
    const float* Wsrc[8] = { (const float*)d_in[4],  (const float*)d_in[6],
                             (const float*)d_in[8],  (const float*)d_in[10],
                             (const float*)d_in[12], (const float*)d_in[14],
                             (const float*)d_in[16], (const float*)d_in[18] };
    const float* sa_bq = (const float*)d_in[5],  *sa_bk = (const float*)d_in[7];
    const float* sa_bv = (const float*)d_in[9],  *sa_bo = (const float*)d_in[11];
    const float* ca_bq = (const float*)d_in[13], *ca_bk = (const float*)d_in[15];
    const float* ca_bv = (const float*)d_in[17], *ca_bo = (const float*)d_in[19];
    const float* ff_W1 = (const float*)d_in[20], *ff_b1 = (const float*)d_in[21];
    const float* ff_W2 = (const float*)d_in[22], *ff_b2 = (const float*)d_in[23];
    const float* ln1_g = (const float*)d_in[24], *ln1_b = (const float*)d_in[25];
    const float* ln2_g = (const float*)d_in[26], *ln2_b = (const float*)d_in[27];
    const float* ln3_g = (const float*)d_in[28], *ln3_b = (const float*)d_in[29];

    const int M = BATCH * TLEN;                     // 4096 rows
    const size_t MB = 1024 * 1024;
    if (ws_size < 96 * MB) return;
    char* w = (char*)d_ws;

    // bf16 workspace layout (96 MB):
    //  [0,16)   : 8 transposed square weights, 2 MB each
    //  [16,24)  : W1T (4096x1024)   [24,32) : W2T (1024x4096)
    //  [32,40)  : xb   [40,48) : encb
    //  [48,96)  : 6 activation slots A0..A5, 8 MB each (H = 32 MB over A0..A3)
    unsigned short* WT[8];
    for (int i = 0; i < 8; ++i) WT[i] = (unsigned short*)(w + (size_t)i * 2 * MB);
    unsigned short* W1T  = (unsigned short*)(w + 16 * MB);
    unsigned short* W2T  = (unsigned short*)(w + 24 * MB);
    unsigned short* xb   = (unsigned short*)(w + 32 * MB);
    unsigned short* encb = (unsigned short*)(w + 40 * MB);
    unsigned short* Aq[6];
    for (int i = 0; i < 6; ++i) Aq[i] = (unsigned short*)(w + (48 + (size_t)8 * i) * MB);

    // ---- one-time bf16 conversion / weight transposition ----
    for (int i = 0; i < 8; ++i)
        tcvt_kernel<<<dim3(16, 16), 256, 0, stream>>>(Wsrc[i], WT[i], D_MODEL, D_MODEL);
    tcvt_kernel<<<dim3(D_FF / 64, D_MODEL / 64), 256, 0, stream>>>(ff_W1, W1T, D_MODEL, D_FF);
    tcvt_kernel<<<dim3(D_MODEL / 64, D_FF / 64), 256, 0, stream>>>(ff_W2, W2T, D_FF, D_MODEL);
    cvt_bf16_kernel<<<(M * D_MODEL) / 2048, 256, 0, stream>>>(x,   xb,   M * D_MODEL);
    cvt_bf16_kernel<<<(M * D_MODEL) / 2048, 256, 0, stream>>>(enc, encb, M * D_MODEL);

    const dim3 gP(D_MODEL / 128, M / 128);          // (8, 32) projection GEMMs
    const dim3 gF1(D_FF / 128,   M / 128);          // (32, 32) FFN up
    const dim3 gA(TLEN / 64, BATCH * N_HEADS);      // (16, 64) attention
    const dim3 gLN(M);

    // ---- self-attention ----
    gemm_bias_kernel<<<gP, 256, 0, stream>>>(xb, WT[0], sa_bq, Aq[0], M, D_MODEL, D_MODEL, 0);
    gemm_bias_kernel<<<gP, 256, 0, stream>>>(xb, WT[1], sa_bk, Aq[1], M, D_MODEL, D_MODEL, 0);
    gemm_bias_kernel<<<gP, 256, 0, stream>>>(xb, WT[2], sa_bv, Aq[2], M, D_MODEL, D_MODEL, 0);
    attn_kernel<<<gA, 128, 0, stream>>>(Aq[0], Aq[1], Aq[2], Aq[3], TLEN, TLEN, 1);
    gemm_bias_kernel<<<gP, 256, 0, stream>>>(Aq[3], WT[3], sa_bo, Aq[0], M, D_MODEL, D_MODEL, 0);
    ln_kernel<<<gLN, 256, 0, stream>>>(xb, Aq[0], ln1_g, ln1_b, Aq[4], 0);   // X1 = A4

    // ---- cross-attention ----
    gemm_bias_kernel<<<gP, 256, 0, stream>>>(Aq[4], WT[4], ca_bq, Aq[0], M, D_MODEL, D_MODEL, 0);
    gemm_bias_kernel<<<gP, 256, 0, stream>>>(encb,  WT[5], ca_bk, Aq[1], M, D_MODEL, D_MODEL, 0);
    gemm_bias_kernel<<<gP, 256, 0, stream>>>(encb,  WT[6], ca_bv, Aq[2], M, D_MODEL, D_MODEL, 0);
    attn_kernel<<<gA, 128, 0, stream>>>(Aq[0], Aq[1], Aq[2], Aq[3], TLEN, SLEN, 0);
    gemm_bias_kernel<<<gP, 256, 0, stream>>>(Aq[3], WT[7], ca_bo, Aq[0], M, D_MODEL, D_MODEL, 0);
    ln_kernel<<<gLN, 256, 0, stream>>>(Aq[4], Aq[0], ln2_g, ln2_b, Aq[5], 0); // X2 = A5

    // ---- FFN (hidden bf16, 32 MB over freed slots A0..A3) ----
    unsigned short* Hbuf = Aq[0];
    gemm_bias_kernel<<<gF1, 256, 0, stream>>>(Aq[5], W1T, ff_b1, Hbuf, M, D_FF, D_MODEL, 1);
    gemm_bias_kernel<<<gP, 256, 0, stream>>>(Hbuf, W2T, ff_b2, Aq[4], M, D_MODEL, D_FF, 0);
    ln_kernel<<<gLN, 256, 0, stream>>>(Aq[5], Aq[4], ln3_g, ln3_b, d_out, 1);
}
